// DLRM_Net_48369921687830
// MI455X (gfx1250) — matrix operand order repaired and checked by reading, not compiled, not run on hardware
//
#include <hip/hip_runtime.h>
#include <hip/hip_bf16.h>

typedef __bf16 bf16_t;
typedef __bf16 v16bf __attribute__((ext_vector_type(16)));
typedef __bf16 v8bf  __attribute__((ext_vector_type(8)));
typedef float  v8f   __attribute__((ext_vector_type(8)));

#define BB    16384      // batch
#define TT    26         // tables
#define NROW  50000      // rows per table
#define DD    128        // embedding dim
#define LL    4          // bag size
#define TCS   4096       // Tcat per-sample stride (32 rows * 128)
#define RW    480        // padded interaction-output width (479 -> 480)

// ---------- bf16 helpers (explicit bit ops; no reliance on __bf16 arithmetic) ----------
static __device__ __forceinline__ float bf2f(bf16_t h) {
  unsigned short s = __builtin_bit_cast(unsigned short, h);
  unsigned u = ((unsigned)s) << 16;
  return __builtin_bit_cast(float, u);
}
static __device__ __forceinline__ unsigned short f2bfu(float f) {
  unsigned u = __builtin_bit_cast(unsigned, f);
  unsigned r = u + 0x7FFFu + ((u >> 16) & 1u);   // round-to-nearest-even
  return (unsigned short)(r >> 16);
}
static __device__ __forceinline__ bf16_t f2bf(float f) {
  unsigned short h = f2bfu(f);
  return __builtin_bit_cast(bf16_t, h);
}

// ---------- WMMA operand loaders (16x16x32 bf16, contiguous k order in A and B) ----------
// A (16x32 MxK): lane holds row m=lane%16; lanes 0-15 hold K kk..kk+15,
// lanes 16-31 hold K kk+16..kk+31.  half8 = (lane&16)?8:0.
static __device__ __forceinline__ v16bf load_a_tile(const bf16_t* __restrict__ row,
                                                    int kk, int half8) {
  return *(const v16bf*)(row + kk + 2 * half8);
}
// B (32x16 KxN): lane holds column n=lane%16; lanes 0-15 hold K kk..kk+15,
// lanes 16-31 hold K kk+16..kk+31 (one contiguous 32-byte load).  half16 = (lane&16)?16:0.
static __device__ __forceinline__ v16bf load_b_tile(const bf16_t* __restrict__ col,
                                                    int kk, int half16) {
  return *(const v16bf*)(col + kk + half16);
}
static __device__ __forceinline__ v8f wmma_bf16(v16bf a, v16bf b, v8f c) {
  return __builtin_amdgcn_wmma_f32_16x16x32_bf16(false, a, false, b, (short)0, c,
                                                 false, false);
}

// ---------- fp32 -> bf16 convert with K padding (weights + dense_x) ----------
__global__ void pad_convert_bf16(const float* __restrict__ src, bf16_t* __restrict__ dst,
                                 int M, int Ksrc, int Kdst) {
  int i = blockIdx.x * blockDim.x + threadIdx.x;
  if (i >= M * Kdst) return;
  int m = i / Kdst, k = i - m * Kdst;
  float v = (k < Ksrc) ? src[(size_t)m * Ksrc + k] : 0.0f;
  dst[i] = f2bf(v);
}

// ---------- zero Tcat pad rows 27..31 (so interaction WMMA keeps EXEC uniform) ----------
__global__ void zero_tcat_pad(bf16_t* __restrict__ tcat) {
  int j = blockIdx.x * blockDim.x + threadIdx.x;  // each thread writes 8 bf16 (16 B)
  int b = j / 80;                                 // 5 rows * 128 / 8 = 80 chunks/sample
  int e = (j - b * 80) * 8;
  uint4 z = {0u, 0u, 0u, 0u};
  *(uint4*)(tcat + (size_t)b * TCS + 27 * DD + e) = z;
}

// ---------- EmbeddingBag sum-pool: one wave per (table, sample) ----------
__global__ void __launch_bounds__(256)
emb_pool(const float* __restrict__ emb, const int* __restrict__ lsi,
         bf16_t* __restrict__ tcat) {
  int w = blockIdx.x * (blockDim.x >> 5) + (threadIdx.x >> 5);
  int lane = threadIdx.x & 31;
  int t = w / BB;
  int b = w - t * BB;
  const int* idx = lsi + ((size_t)t * BB + b) * LL;
  const float* tbl = emb + (size_t)t * NROW * DD;
  float4 acc = {0.f, 0.f, 0.f, 0.f};
#pragma unroll
  for (int i = 0; i < LL; ++i) {
    const float4* p = (const float4*)(tbl + (size_t)idx[i] * DD) + lane;  // 512B coalesced/wave
    float4 v = *p;
    acc.x += v.x; acc.y += v.y; acc.z += v.z; acc.w += v.w;
  }
  unsigned long long pk =
      (unsigned long long)f2bfu(acc.x) |
      ((unsigned long long)f2bfu(acc.y) << 16) |
      ((unsigned long long)f2bfu(acc.z) << 32) |
      ((unsigned long long)f2bfu(acc.w) << 48);
  *(unsigned long long*)(tcat + (size_t)b * TCS + (1 + t) * DD + lane * 4) = pk;
}

// ---------- generic bf16 WMMA GEMM: Out[m,n] = act(A[m,:]·W[n,:] + bias[n]) ----------
// A: [M,K] bf16 (lda=K), W: [N,K] bf16 row-major (= W.T GEMM), bias fp32, Out bf16 (ldo).
// One wave computes a 16(M) x 64(N) tile; K must be a multiple of 32.
__global__ void __launch_bounds__(256)
gemm_bf16_wmma(const bf16_t* __restrict__ A, const bf16_t* __restrict__ W,
               const float* __restrict__ bias, bf16_t* __restrict__ Out,
               int N, int K, int ldo) {
  int wave = blockIdx.x * (blockDim.x >> 5) + (threadIdx.x >> 5);
  int lane = threadIdx.x & 31;
  int ntiles = N >> 6;
  int mi = wave / ntiles;
  int ni = wave - mi * ntiles;
  int half8 = (lane & 16) ? 8 : 0;
  int half16 = (lane & 16) ? 16 : 0;
  const bf16_t* arow = A + (size_t)((mi << 4) + (lane & 15)) * K;
  const bf16_t* wcol = W + (size_t)((ni << 6) + (lane & 15)) * K;
  v8f acc0 = {}, acc1 = {}, acc2 = {}, acc3 = {};
  for (int kk = 0; kk < K; kk += 32) {
    v16bf a  = load_a_tile(arow, kk, half8);
    v16bf b0 = load_b_tile(wcol, kk, half16);
    v16bf b1 = load_b_tile(wcol + (size_t)16 * K, kk, half16);
    v16bf b2 = load_b_tile(wcol + (size_t)32 * K, kk, half16);
    v16bf b3 = load_b_tile(wcol + (size_t)48 * K, kk, half16);
    acc0 = wmma_bf16(a, b0, acc0);
    acc1 = wmma_bf16(a, b1, acc1);
    acc2 = wmma_bf16(a, b2, acc2);
    acc3 = wmma_bf16(a, b3, acc3);
  }
  // C/D layout: lane holds col n=lane%16; VGPR v holds row (v + 8*(lane/16)).
  int col0 = (ni << 6) + (lane & 15);
  int mrow0 = (mi << 4) + ((lane & 16) ? 8 : 0);
  v8f accs[4] = {acc0, acc1, acc2, acc3};
#pragma unroll
  for (int tsub = 0; tsub < 4; ++tsub) {
    int n = col0 + tsub * 16;
    float bv = bias[n];
#pragma unroll
    for (int v = 0; v < 8; ++v) {
      float val = accs[tsub][v] + bv;
      val = val > 0.f ? val : 0.f;  // ReLU (all WMMA layers are ReLU; sigmoid in final_dot)
      Out[(size_t)(mrow0 + v) * ldo + n] = f2bf(val);
    }
  }
}

// ---------- per-sample 27x27 Gram interaction via WMMA; build R[B,480] ----------
__global__ void __launch_bounds__(256)
interact_wmma(const bf16_t* __restrict__ tcat, bf16_t* __restrict__ R) {
  int w = blockIdx.x * (blockDim.x >> 5) + (threadIdx.x >> 5);  // sample id
  int lane = threadIdx.x & 31;
  const bf16_t* base = tcat + (size_t)w * TCS;
  int half8 = (lane & 16) ? 8 : 0;
  int half16 = (lane & 16) ? 16 : 0;
  const bf16_t* r0 = base + (size_t)(lane & 15) * DD;         // rows 0..15
  const bf16_t* r1 = base + (size_t)(16 + (lane & 15)) * DD;  // rows 16..31 (27..31 zeroed)
  v8f z00 = {}, z10 = {}, z11 = {};
#pragma unroll
  for (int kk = 0; kk < DD; kk += 32) {
    v16bf a0 = load_a_tile(r0, kk, half8);
    v16bf a1 = load_a_tile(r1, kk, half8);
    v16bf b0 = load_b_tile(r0, kk, half16);
    v16bf b1 = load_b_tile(r1, kk, half16);
    z00 = wmma_bf16(a0, b0, z00);
    z10 = wmma_bf16(a1, b0, z10);
    z11 = wmma_bf16(a1, b1, z11);
  }
  bf16_t* Rrow = R + (size_t)w * RW;
  // R[:, 0:128] = x  (Tcat row 0), 4 bf16 per lane
  *(unsigned long long*)(Rrow + (lane << 2)) =
      *(const unsigned long long*)(base + (lane << 2));
  if (lane == 0) Rrow[RW - 1] = f2bf(0.f);  // pad element 479
  // scatter strict-lower-triangle entries: p = i*(i-1)/2 + j  (i>j)
  int n = lane & 15;
  int hi = (lane & 16) ? 8 : 0;
#pragma unroll
  for (int v = 0; v < 8; ++v) {
    int m = hi + v;
    if (m > n) {                                     // tile (0,0): i=m, j=n
      Rrow[128 + ((m * (m - 1)) >> 1) + n] = f2bf(z00[v]);
    }
    {                                                // tile (1,0): i=16+m, j=n
      int i = 16 + m;
      if (i < 27) Rrow[128 + ((i * (i - 1)) >> 1) + n] = f2bf(z10[v]);
    }
    {                                                // tile (1,1): i=16+m, j=16+n
      int i = 16 + m, j = 16 + n;
      if (i < 27 && j < 27 && i > j)
        Rrow[128 + ((i * (i - 1)) >> 1) + j] = f2bf(z11[v]);
    }
  }
}

// ---------- final 256 -> 1 layer + sigmoid: one wave per sample ----------
__global__ void __launch_bounds__(256)
final_dot(const bf16_t* __restrict__ H, const float* __restrict__ W5,
          const float* __restrict__ b5, float* __restrict__ out) {
  int w = blockIdx.x * (blockDim.x >> 5) + (threadIdx.x >> 5);
  int lane = threadIdx.x & 31;
  const bf16_t* h = H + (size_t)w * 256;
  float s = 0.f;
#pragma unroll
  for (int i = 0; i < 8; ++i) {
    int k = lane * 8 + i;
    s += bf2f(h[k]) * W5[k];
  }
#pragma unroll
  for (int off = 16; off > 0; off >>= 1) s += __shfl_xor(s, off, 32);
  if (lane == 0) {
    float y = s + b5[0];
    out[w] = 1.f / (1.f + __expf(-y));
  }
}

extern "C" void kernel_launch(void* const* d_in, const int* in_sizes, int n_in,
                              void* d_out, int out_size, void* d_ws, size_t ws_size,
                              hipStream_t stream) {
  (void)in_sizes; (void)n_in; (void)out_size; (void)ws_size;
  const float* dense_x = (const float*)d_in[0];
  const int*   lsi     = (const int*)d_in[1];     // indices (int32 per harness contract)
  const float* emb     = (const float*)d_in[2];
  const float* Wb1 = (const float*)d_in[3];  const float* bb1 = (const float*)d_in[4];
  const float* Wb2 = (const float*)d_in[5];  const float* bb2 = (const float*)d_in[6];
  const float* Wb3 = (const float*)d_in[7];  const float* bb3 = (const float*)d_in[8];
  const float* Wt1 = (const float*)d_in[9];  const float* bt1 = (const float*)d_in[10];
  const float* Wt2 = (const float*)d_in[11]; const float* bt2 = (const float*)d_in[12];
  const float* Wt3 = (const float*)d_in[13]; const float* bt3 = (const float*)d_in[14];
  const float* Wt4 = (const float*)d_in[15]; const float* bt4 = (const float*)d_in[16];
  const float* Wt5 = (const float*)d_in[17]; const float* bt5 = (const float*)d_in[18];
  float* out = (float*)d_out;

  char* ws = (char*)d_ws;
  size_t off = 0;
  auto take = [&](size_t bytes) -> void* {
    void* p = ws + off;
    off += (bytes + 255) & ~((size_t)255);
    return p;
  };
  bf16_t* X0p  = (bf16_t*)take((size_t)BB * 32 * 2);     // padded bf16 dense_x [B,32]
  bf16_t* W1p  = (bf16_t*)take((size_t)512 * 32 * 2);
  bf16_t* W2c  = (bf16_t*)take((size_t)256 * 512 * 2);
  bf16_t* W3c  = (bf16_t*)take((size_t)128 * 256 * 2);
  bf16_t* T1p  = (bf16_t*)take((size_t)1024 * RW * 2);
  bf16_t* T2c  = (bf16_t*)take((size_t)1024 * 1024 * 2);
  bf16_t* T3c  = (bf16_t*)take((size_t)512 * 1024 * 2);
  bf16_t* T4c  = (bf16_t*)take((size_t)256 * 512 * 2);
  bf16_t* A1   = (bf16_t*)take((size_t)BB * 512 * 2);
  bf16_t* A2   = (bf16_t*)take((size_t)BB * 256 * 2);
  bf16_t* Tcat = (bf16_t*)take((size_t)BB * TCS * 2);    // [B,32,128]; reused for H1..H4
  bf16_t* R    = (bf16_t*)take((size_t)BB * RW * 2);
  // Top-MLP activations alias the (dead after interaction) Tcat region (92.2MB <= 134MB).
  bf16_t* H1 = Tcat;
  bf16_t* H2 = H1 + (size_t)BB * 1024;
  bf16_t* H3 = H2 + (size_t)BB * 1024;
  bf16_t* H4 = H3 + (size_t)BB * 512;

  auto cdiv = [](int a, int b) { return (a + b - 1) / b; };

  // fp32 -> bf16 weight/input conversion (K padded to multiples of 32)
  pad_convert_bf16<<<cdiv(BB * 32, 256), 256, 0, stream>>>(dense_x, X0p, BB, 13, 32);
  pad_convert_bf16<<<cdiv(512 * 32, 256), 256, 0, stream>>>(Wb1, W1p, 512, 13, 32);
  pad_convert_bf16<<<cdiv(256 * 512, 256), 256, 0, stream>>>(Wb2, W2c, 256, 512, 512);
  pad_convert_bf16<<<cdiv(128 * 256, 256), 256, 0, stream>>>(Wb3, W3c, 128, 256, 256);
  pad_convert_bf16<<<cdiv(1024 * RW, 256), 256, 0, stream>>>(Wt1, T1p, 1024, 479, RW);
  pad_convert_bf16<<<cdiv(1024 * 1024, 256), 256, 0, stream>>>(Wt2, T2c, 1024, 1024, 1024);
  pad_convert_bf16<<<cdiv(512 * 1024, 256), 256, 0, stream>>>(Wt3, T3c, 512, 1024, 1024);
  pad_convert_bf16<<<cdiv(256 * 512, 256), 256, 0, stream>>>(Wt4, T4c, 256, 512, 512);

  // Tcat pad rows + embedding pooling (memory-bound roofline term: ~0.87 GB gather)
  zero_tcat_pad<<<BB * 80 / 256, 256, 0, stream>>>(Tcat);
  emb_pool<<<TT * BB / 8, 256, 0, stream>>>(emb, lsi, Tcat);

  // Bottom MLP: 13(->32) -> 512 -> 256 -> 128 (layer-3 writes Tcat row 0, stride TCS)
  gemm_bf16_wmma<<<(BB / 16) * (512 / 64) / 8, 256, 0, stream>>>(X0p, W1p, bb1, A1, 512, 32, 512);
  gemm_bf16_wmma<<<(BB / 16) * (256 / 64) / 8, 256, 0, stream>>>(A1, W2c, bb2, A2, 256, 512, 256);
  gemm_bf16_wmma<<<(BB / 16) * (128 / 64) / 8, 256, 0, stream>>>(A2, W3c, bb3, Tcat, 128, 256, TCS);

  // Dot-feature interaction (per-sample 27x27 Gram via 3 WMMA tiles) -> R[B,480]
  interact_wmma<<<BB / 8, 256, 0, stream>>>(Tcat, R);

  // Top MLP: 479(->480) -> 1024 -> 1024 -> 512 -> 256, then 256 -> 1 + sigmoid
  gemm_bf16_wmma<<<(BB / 16) * (1024 / 64) / 8, 256, 0, stream>>>(R, T1p, bt1, H1, 1024, RW, 1024);
  gemm_bf16_wmma<<<(BB / 16) * (1024 / 64) / 8, 256, 0, stream>>>(H1, T2c, bt2, H2, 1024, 1024, 1024);
  gemm_bf16_wmma<<<(BB / 16) * (512 / 64) / 8, 256, 0, stream>>>(H2, T3c, bt3, H3, 512, 1024, 512);
  gemm_bf16_wmma<<<(BB / 16) * (256 / 64) / 8, 256, 0, stream>>>(H3, T4c, bt4, H4, 256, 512, 256);
  final_dot<<<BB / 8, 256, 0, stream>>>(H4, Wt5, bt5, out);
}
